// GQAttention_20375324852545
// MI455X (gfx1250) — matrix operand order repaired and checked by reading, not compiled, not run on hardware
//
#include <hip/hip_runtime.h>

// ---------------------------------------------------------------------------
// GQA attention for MI455X (gfx1250), bf16 WMMA + TDM async-tensor staging.
//   B=4, S=2048, D=2048, NH=16, HD=128, NKV=4, DKV=512
//   head h uses KV group (h % 4)  [derived from repeat-on-sequence-axis quirk]
// ---------------------------------------------------------------------------

#define B_   4
#define S_   2048
#define D_   2048
#define NH_  16
#define HD_  128
#define NKV_ 4
#define DKV_ 512
#define BS_  (B_ * S_)           // 8192 rows for projections
#define SCALE_ 0.08838834764831845f   // 1/sqrt(128)
#define EPS_   1e-9f

typedef __attribute__((ext_vector_type(16))) __bf16 v16bf;
typedef __attribute__((ext_vector_type(8)))  __bf16 v8bf;
typedef __attribute__((ext_vector_type(4)))  __bf16 v4bf;
typedef __attribute__((ext_vector_type(8)))  float  v8f;
typedef unsigned int u32x4 __attribute__((ext_vector_type(4)));
typedef int          i32x4 __attribute__((ext_vector_type(4)));
typedef int          i32x8 __attribute__((ext_vector_type(8)));

union V16U { v16bf v; v8bf h[2]; };

static __device__ __forceinline__ v8f wmma_bf16(v16bf a, v16bf b, v8f c) {
  // 8 args: (neg_a, A, neg_b, B, c_mod, C, reuse_a, reuse_b)
  return __builtin_amdgcn_wmma_f32_16x16x32_bf16(false, a, false, b,
                                                 (short)0, c, false, false);
}

// Raw LDS byte offset of a __shared__ object (generic -> addrspace(3) -> int).
static __device__ __forceinline__ unsigned lds_addr_of(const void* p) {
  return (unsigned)(unsigned long long)
         (__attribute__((address_space(3))) const char*)(const char*)p;
}

// Toolchain-dependent TDM builtin arity (see CDNA5_HIP.md):
//   ROCm 7.2 clang-22:        5 args (g0, g1, g2, g3, cpol)
//   amdgpu-toolchain clang-23: 6 args (g0, g1, g2, g3, g4, cpol)
#if __has_include(<hip/amd_detail/amd_gfx1250_TDM.h>)
#define TDM_LOAD(g0, g1, g2, g3) \
  __builtin_amdgcn_tensor_load_to_lds((g0), (g1), (g2), (g3), \
                                      (i32x8){0,0,0,0,0,0,0,0}, 0)
#else
#define TDM_LOAD(g0, g1, g2, g3) \
  __builtin_amdgcn_tensor_load_to_lds((g0), (g1), (g2), (g3), 0)
#endif

// Issue a TDM load of a 2-D bf16 tile (rows x row_elems) from global memory
// (row stride src_stride_elems) into LDS at lds_byte_off, inserting
// pad_amount DWORDs of LDS padding every pad_interval-coded DWORDs.
// D# packing per CDNA5 ISA.
static __device__ __forceinline__ void tdm_load_2d_bf16(
    unsigned lds_byte_off, const void* gaddr, unsigned rows, unsigned row_elems,
    unsigned src_stride_elems, unsigned pad_interval_code, unsigned pad_amount_code)
{
  const unsigned long long ga = (unsigned long long)gaddr;
  u32x4 g0;
  g0[0] = 1u;                                          // count=1 (valid D#)
  g0[1] = lds_byte_off;                                // lds_addr [63:32]
  g0[2] = (unsigned)(ga & 0xFFFFFFFFu);                // global_addr lo
  g0[3] = (unsigned)((ga >> 32) & 0x01FFFFFFu)         // global_addr hi
        | (2u << 30);                                  // type=2 ("image")
  i32x8 g1;
  g1[0] = (int)((1u << 16)                             // data_size=1 -> 2B
              | (1u << 20)                             // pad_enable
              | (pad_interval_code << 22)
              | (pad_amount_code << 25));              // workgroup_mask=0
  g1[1] = (int)((row_elems & 0xFFFFu) << 16);          // tensor_dim0 [63:48]
  g1[2] = (int)(((row_elems >> 16) & 0xFFFFu)          // tensor_dim0 [79:64]
              | ((rows & 0xFFFFu) << 16));             // tensor_dim1 [95:80]
  g1[3] = (int)(((rows >> 16) & 0xFFFFu)               // tensor_dim1 [111:96]
              | ((row_elems & 0xFFFFu) << 16));        // tile_dim0  [127:112]
  g1[4] = (int)(rows & 0xFFFFu);                       // tile_dim1; tile_dim2=0
  g1[5] = (int)src_stride_elems;                       // tensor_dim0_stride lo
  g1[6] = 0;                                           // stride hi / dim1_stride
  g1[7] = 0;
  const i32x4 z4 = {0, 0, 0, 0};                       // groups 2/3 unused (2-D)
  TDM_LOAD(g0, g1, z4, z4);
}

// ---------------------------------------------------------------------------
// GEMM: C(bf16, MxN) = A(f32, MxK) @ W(f32, KxN) + bias(f32, N)
// block tile 128x128, K-step 32, 256 threads = 8 waves, wave tile 32x64.
// (fp32 sources need on-the-fly bf16 conversion, so staging stays VALU-based.)
// ---------------------------------------------------------------------------
__global__ __launch_bounds__(256)
void proj_gemm_bf16(const float* __restrict__ A, const float* __restrict__ W,
                    const float* __restrict__ bias, __bf16* __restrict__ C,
                    int M, int N, int K) {
  __shared__ __bf16 Asm[128][40];   // 128 rows x 32 K (80B stride: aligned, bank-spread)
  __shared__ __bf16 BTsm[128][40];  // W tile transposed: 128 n-rows x 32 K

  const int tid  = threadIdx.x;
  const int lane = tid & 31;
  const int wave = tid >> 5;
  const int m0 = blockIdx.y * 128;
  const int n0 = blockIdx.x * 128;
  const int mw = (wave & 3) * 32;
  const int nw = (wave >> 2) * 64;
  const int lc = lane & 15, half = lane >> 4;

  v8f acc[2][4];
  #pragma unroll
  for (int mt = 0; mt < 2; ++mt)
    #pragma unroll
    for (int nt = 0; nt < 4; ++nt) acc[mt][nt] = {};

  for (int k0 = 0; k0 < K; k0 += 32) {
    // ---- stage A tile: 128x32 f32 -> bf16 (vectorized, coalesced) ----
    #pragma unroll
    for (int i = 0; i < 4; ++i) {
      int c   = i * 256 + tid;          // 1024 float4 chunks
      int row = c >> 3, kq = (c & 7) << 2;
      const float4 f = *(const float4*)&A[(size_t)(m0 + row) * K + k0 + kq];
      v4bf hbits;
      hbits[0] = (__bf16)f.x; hbits[1] = (__bf16)f.y;
      hbits[2] = (__bf16)f.z; hbits[3] = (__bf16)f.w;
      *(v4bf*)&Asm[row][kq] = hbits;
    }
    // ---- stage W tile transposed: read (32 k)x(128 n) coalesced ----
    #pragma unroll
    for (int i = 0; i < 16; ++i) {
      int e  = i * 256 + tid;           // 4096 elements
      int kr = e >> 7, nc = e & 127;
      BTsm[nc][kr] = (__bf16)W[(size_t)(k0 + kr) * N + n0 + nc];
    }
    __syncthreads();

    // ---- A fragments (16x32 each), two M tiles ----
    v16bf afr[2];
    {
      int rowA = mw + lc;
      int koff = half * 8;
      #pragma unroll
      for (int mt = 0; mt < 2; ++mt) {
        V16U u;
        u.h[0] = *(const v8bf*)&Asm[rowA + mt * 16][koff];
        u.h[1] = *(const v8bf*)&Asm[rowA + mt * 16][koff + 16];
        afr[mt] = u.v;
      }
    }
    // ---- B fragments + WMMA ----
    #pragma unroll
    for (int nt = 0; nt < 4; ++nt) {
      int nrow = nw + nt * 16 + lc;
      int kb   = half * 8;
      V16U u;
      u.h[0] = *(const v8bf*)&BTsm[nrow][kb];
      u.h[1] = *(const v8bf*)&BTsm[nrow][kb + 16];
      acc[0][nt] = wmma_bf16(afr[0], u.v, acc[0][nt]);
      acc[1][nt] = wmma_bf16(afr[1], u.v, acc[1][nt]);
    }
    __syncthreads();
  }

  // ---- epilogue: +bias, convert to bf16, store ----
  #pragma unroll
  for (int nt = 0; nt < 4; ++nt) {
    int col  = n0 + nw + nt * 16 + lc;
    float bb = bias[col];
    #pragma unroll
    for (int mt = 0; mt < 2; ++mt)
      #pragma unroll
      for (int r = 0; r < 8; ++r) {
        int row = m0 + mw + mt * 16 + r + 8 * half;   // C/D VGPR layout
        C[(size_t)row * N + col] = (__bf16)(acc[mt][nt][r] + bb);
      }
  }
}

// ---------------------------------------------------------------------------
// Flash attention: block = 64 query rows of one (b,h). 4 waves x 16 rows.
// Q tile and K chunks staged by the Tensor Data Mover (pad 4 DWORDs / 256B
// -> 272B LDS row stride); V staged manually (needs transpose).
// ---------------------------------------------------------------------------
__global__ __launch_bounds__(128)
void gqa_flash_attn(const __bf16* __restrict__ QH, const __bf16* __restrict__ KH,
                    const __bf16* __restrict__ VH, float* __restrict__ out) {
  __shared__ __bf16 Qsm[64][136];      // 64 q-rows x 128 d (TDM-padded)
  __shared__ __bf16 Ksm[64][136];      // 64 keys x 128 d  (TDM-padded)
  __shared__ __bf16 VTsm[128][72];     // V transposed: 128 d x 64 keys
  __shared__ __bf16 Psm[4][16][72];    // per-wave P relayout buffer

  const int tid  = threadIdx.x;
  const int lane = tid & 31;
  const int wave = tid >> 5;
  const int lc = lane & 15, half = lane >> 4;

  const int q0 = blockIdx.x * 64;
  const int h  = blockIdx.y;
  const int b  = blockIdx.z;
  const int g  = h & (NKV_ - 1);       // head h -> KV group h % 4

  const size_t qbase  = ((size_t)b * S_ + q0) * D_ + (size_t)h * HD_;
  const size_t kvbase = ((size_t)b * S_) * DKV_ + (size_t)g * HD_;

  // ---- stage Q tile via TDM (wave 0 issues; pad codes: 64 DW interval=5,
  //      4 DW amount=3 -> 272B rows matching Qsm stride 136) ----
  if (wave == 0) {
    tdm_load_2d_bf16(lds_addr_of(&Qsm[0][0]), &QH[qbase],
                     /*rows=*/64, /*row_elems=*/HD_, /*stride=*/D_,
                     /*pad_interval=*/5, /*pad_amount=*/3);
    __builtin_amdgcn_s_wait_tensorcnt(0);
  }
  __syncthreads();

  // ---- persistent Q A-fragments: 4 K-steps covering HD=128 ----
  v16bf qfr[4];
  {
    int row  = wave * 16 + lc;
    int koff = half * 8;
    #pragma unroll
    for (int d = 0; d < 4; ++d) {
      V16U u;
      u.h[0] = *(const v8bf*)&Qsm[row][d * 32 + koff];
      u.h[1] = *(const v8bf*)&Qsm[row][d * 32 + koff + 16];
      qfr[d] = u.v;
    }
  }

  v8f Oacc[8];
  #pragma unroll
  for (int nt = 0; nt < 8; ++nt) Oacc[nt] = {};
  float m8[8], l8[8];
  #pragma unroll
  for (int r = 0; r < 8; ++r) { m8[r] = -__builtin_inff(); l8[r] = 0.f; }

  for (int k0 = 0; k0 < S_; k0 += 64) {
    if (k0 + 64 < S_) {  // uniform branch: prefetch next V chunk
      size_t pr = kvbase + (size_t)(k0 + 64 + (tid >> 1)) * DKV_ + (size_t)(tid & 1) * 64;
      __builtin_prefetch(&VH[pr], 0, 0);
    }
    __syncthreads();     // previous chunk fully consumed
    // ---- K chunk via TDM (async; overlaps the manual V staging below) ----
    if (wave == 0) {
      tdm_load_2d_bf16(lds_addr_of(&Ksm[0][0]),
                       &KH[kvbase + (size_t)k0 * DKV_],
                       /*rows=*/64, /*row_elems=*/HD_, /*stride=*/DKV_,
                       /*pad_interval=*/5, /*pad_amount=*/3);
    }
    // ---- stage V chunk transposed (all threads) ----
    #pragma unroll
    for (int i = 0; i < 8; ++i) {
      int e = i * 128 + tid;
      int key = e >> 4, d8 = (e & 15) << 3;
      v8bf vv = *(const v8bf*)&VH[kvbase + (size_t)(k0 + key) * DKV_ + d8];
      #pragma unroll
      for (int j = 0; j < 8; ++j) VTsm[d8 + j][key] = vv[j];
    }
    if (wave == 0) __builtin_amdgcn_s_wait_tensorcnt(0);
    __syncthreads();

    // ---- scores S = Q @ K^T  (16 WMMAs) ----
    v8f Sacc[4];
    #pragma unroll
    for (int nt = 0; nt < 4; ++nt) Sacc[nt] = {};
    #pragma unroll
    for (int nt = 0; nt < 4; ++nt) {
      int key = nt * 16 + lc;
      #pragma unroll
      for (int d = 0; d < 4; ++d) {
        int db = d * 32 + half * 8;
        V16U u;
        u.h[0] = *(const v8bf*)&Ksm[key][db];
        u.h[1] = *(const v8bf*)&Ksm[key][db + 16];
        Sacc[nt] = wmma_bf16(qfr[d], u.v, Sacc[nt]);
      }
    }

    // ---- online softmax (row stats live per (r, lane-half)) ----
    float corr[8];
    #pragma unroll
    for (int r = 0; r < 8; ++r) {
      float v = fmaxf(fmaxf(Sacc[0][r], Sacc[1][r]), fmaxf(Sacc[2][r], Sacc[3][r]));
      v = fmaxf(v, __shfl_xor(v, 1, 32));
      v = fmaxf(v, __shfl_xor(v, 2, 32));
      v = fmaxf(v, __shfl_xor(v, 4, 32));
      v = fmaxf(v, __shfl_xor(v, 8, 32));      // stays within 16-lane half
      float mn = fmaxf(m8[r], v * SCALE_);
      corr[r] = __expf(m8[r] - mn);
      m8[r] = mn;
    }
    #pragma unroll
    for (int r = 0; r < 8; ++r) {
      float s = 0.f;
      int prow = r + 8 * half;
      #pragma unroll
      for (int nt = 0; nt < 4; ++nt) {
        float p = __expf(Sacc[nt][r] * SCALE_ - m8[r]);
        s += p;
        Psm[wave][prow][nt * 16 + lc] = (__bf16)p;   // D-layout -> LDS
      }
      s += __shfl_xor(s, 1, 32);
      s += __shfl_xor(s, 2, 32);
      s += __shfl_xor(s, 4, 32);
      s += __shfl_xor(s, 8, 32);
      l8[r] = l8[r] * corr[r] + s;
    }
    #pragma unroll
    for (int nt = 0; nt < 8; ++nt)
      #pragma unroll
      for (int r = 0; r < 8; ++r) Oacc[nt][r] *= corr[r];

    // ---- reload P as A-fragments (wave-local LDS, in-order DS ops) ----
    v16bf pfr[2];
    {
      int koff = half * 8;
      #pragma unroll
      for (int kk = 0; kk < 2; ++kk) {
        V16U u;
        u.h[0] = *(const v8bf*)&Psm[wave][lc][kk * 32 + koff];
        u.h[1] = *(const v8bf*)&Psm[wave][lc][kk * 32 + koff + 16];
        pfr[kk] = u.v;
      }
    }
    // ---- O += P @ V  (16 WMMAs) ----
    #pragma unroll
    for (int nt = 0; nt < 8; ++nt) {
      int dcol = nt * 16 + lc;
      #pragma unroll
      for (int kk = 0; kk < 2; ++kk) {
        int kb = kk * 32 + half * 8;
        V16U u;
        u.h[0] = *(const v8bf*)&VTsm[dcol][kb];
        u.h[1] = *(const v8bf*)&VTsm[dcol][kb + 16];
        Oacc[nt] = wmma_bf16(pfr[kk], u.v, Oacc[nt]);
      }
    }
  }

  // ---- normalize by (l + EPS), store fp32 ----
  #pragma unroll
  for (int r = 0; r < 8; ++r) {
    float inv = 1.f / (l8[r] + EPS_);
    int grow = q0 + wave * 16 + r + 8 * half;
    size_t obase = ((size_t)b * S_ + grow) * D_ + (size_t)h * HD_;
    #pragma unroll
    for (int nt = 0; nt < 8; ++nt)
      out[obase + nt * 16 + lc] = Oacc[nt][r] * inv;
  }
}

// ---------------------------------------------------------------------------
// Launch: 3 projection GEMMs into bf16 workspace, then flash attention.
// Workspace: QH 32MB + KH 8MB + VH 8MB = ~48MB bf16.
// ---------------------------------------------------------------------------
extern "C" void kernel_launch(void* const* d_in, const int* in_sizes, int n_in,
                              void* d_out, int out_size, void* d_ws, size_t ws_size,
                              hipStream_t stream) {
  const float* q  = (const float*)d_in[0];
  const float* k  = (const float*)d_in[1];
  const float* v  = (const float*)d_in[2];
  const float* Wq = (const float*)d_in[3];
  const float* bq = (const float*)d_in[4];
  const float* Wk = (const float*)d_in[5];
  const float* bk = (const float*)d_in[6];
  const float* Wv = (const float*)d_in[7];
  const float* bv = (const float*)d_in[8];
  float* out = (float*)d_out;

  char* ws = (char*)d_ws;
  __bf16* QH = (__bf16*)ws;
  __bf16* KH = (__bf16*)(ws + (size_t)BS_ * D_ * 2);
  __bf16* VH = (__bf16*)(ws + (size_t)BS_ * D_ * 2 + (size_t)BS_ * DKV_ * 2);

  dim3 blk(256);
  proj_gemm_bf16<<<dim3(D_ / 128, BS_ / 128), blk, 0, stream>>>(q, Wq, bq, QH, BS_, D_, D_);
  proj_gemm_bf16<<<dim3(DKV_ / 128, BS_ / 128), blk, 0, stream>>>(k, Wk, bk, KH, BS_, DKV_, D_);
  proj_gemm_bf16<<<dim3(DKV_ / 128, BS_ / 128), blk, 0, stream>>>(v, Wv, bv, VH, BS_, DKV_, D_);

  gqa_flash_attn<<<dim3(S_ / 64, NH_, B_), dim3(128), 0, stream>>>(QH, KH, VH, out);
}
